// GCN_77584289234976
// MI455X (gfx1250) — matrix operand order repaired and checked by reading, not compiled, not run on hardware
//
#include <hip/hip_runtime.h>

// MI455X / gfx1250 implementation.
// Heavy math on v_wmma_f32_16x16x32_bf16 (bf16 in, f32 accum): this problem is
// compute-bound (~48 GFLOP vs <200MB traffic, which fits in the 192MB L2), and
// bf16 WMMA has 8x the K-depth per instruction of the fp32 16x16x4 WMMA.

typedef __attribute__((ext_vector_type(16))) __bf16 v16bf;
typedef __attribute__((ext_vector_type(8)))  __bf16 v8bf;
typedef __attribute__((ext_vector_type(8)))  float  v8f;

#define BB    32      // batch
#define NN    2048    // nodes
#define DD    128     // d_in = d_out
#define EE    64      // embed
#define KK2   256     // 2*DD   (k,i) flattened
#define NJ    32768   // DD*KK2 (o,k,i) flattened
#define CHUNK 256     // nodes per weight-gen chunk

__device__ __forceinline__ v8f wmma_bf16(v16bf a, v16bf b, v8f c) {
  // 8 args: (neg_a, A, neg_b, B, c_mod, C, reuse_a, reuse_b)
  return __builtin_amdgcn_wmma_f32_16x16x32_bf16(false, a, false, b, (short)0, c,
                                                 false, false);
}

// A operand (16x32 bf16), row-major source with leading dim lda.
// ISA layout: lane(hi,r): elems 0..7 = K[k0+8*hi .. +7], elems 8..15 = K[k0+16+8*hi .. +7]
__device__ __forceinline__ v16bf load_A16(const __bf16* base, int lda, int k0, int lane) {
  int r = lane & 15, hi = lane >> 4;
  const __bf16* p = base + (size_t)r * lda + k0 + hi * 8;
  v8bf lo = *(const v8bf*)p;
  v8bf up = *(const v8bf*)(p + 16);
  v16bf a;
#pragma unroll
  for (int i = 0; i < 8; ++i) { a[i] = lo[i]; a[i + 8] = up[i]; }
  return a;
}

// B operand (32x16 bf16) loaded from B^T stored row-major [Ncols][K], leading dim ldb.
// ISA layout: lane(hi,c): elems 0..7 = K[k0+8*hi .. +7], elems 8..15 = K[k0+16+8*hi .. +7] of column c.
__device__ __forceinline__ v16bf load_Bt16(const __bf16* btBase, int ldb, int k0, int lane) {
  int c = lane & 15, hi = lane >> 4;
  const __bf16* p = btBase + (size_t)c * ldb + k0 + hi * 8;
  v8bf lo = *(const v8bf*)p;
  v8bf up = *(const v8bf*)(p + 16);
  v16bf b;
#pragma unroll
  for (int i = 0; i < 8; ++i) { b[i] = lo[i]; b[i + 8] = up[i]; }
  return b;
}

// C/D f32 16x16: VGPR j -> row j + 8*hi, col = lane&15
__device__ __forceinline__ void store_C_bf16(__bf16* base, size_t ldc, v8f c, int lane) {
  int col = lane & 15, hi = lane >> 4;
#pragma unroll
  for (int j = 0; j < 8; ++j) base[(size_t)(j + hi * 8) * ldc + col] = (__bf16)c[j];
}

// ---------------- Kernel 1: e = LayerNorm(node_emb + time_emb), f32 + bf16 ----
__global__ __launch_bounds__(256) void k_prep_e(const float* __restrict__ nodee,
                                                const float* __restrict__ timee,
                                                const float* __restrict__ lnw,
                                                const float* __restrict__ lnb,
                                                float* __restrict__ e32,
                                                __bf16* __restrict__ e16) {
  int lane = threadIdx.x & 31, wave = threadIdx.x >> 5;
  int row = blockIdx.x * 8 + wave;            // 256 blocks * 8 waves = 2048 rows
  const float* src = nodee + (size_t)row * EE;
  float v0 = src[2 * lane]     + timee[2 * lane];
  float v1 = src[2 * lane + 1] + timee[2 * lane + 1];
  float s = v0 + v1;
#pragma unroll
  for (int off = 16; off; off >>= 1) s += __shfl_xor(s, off, 32);
  float mu = s * (1.0f / EE);
  float d0 = v0 - mu, d1 = v1 - mu;
  float q = d0 * d0 + d1 * d1;
#pragma unroll
  for (int off = 16; off; off >>= 1) q += __shfl_xor(q, off, 32);
  float rstd = rsqrtf(q * (1.0f / EE) + 1e-12f);
  float o0 = d0 * rstd * lnw[2 * lane]     + lnb[2 * lane];
  float o1 = d1 * rstd * lnw[2 * lane + 1] + lnb[2 * lane + 1];
  size_t base = (size_t)row * EE + 2 * lane;
  e32[base] = o0;         e32[base + 1] = o1;
  e16[base] = (__bf16)o0; e16[base + 1] = (__bf16)o1;
}

// ---------------- Kernel 2: pack x -> bf16 row-major + per-batch transpose ----
__global__ __launch_bounds__(256) void k_pack_x(const float* __restrict__ x,
                                                __bf16* __restrict__ x16,
                                                __bf16* __restrict__ xT16) {
  size_t idx = (size_t)blockIdx.x * blockDim.x + threadIdx.x;  // BB*NN*DD
  int b = (int)(idx / (NN * DD));
  int rem = (int)(idx % (NN * DD));
  int n = rem / DD, c = rem % DD;
  __bf16 v = (__bf16)x[idx];
  x16[idx] = v;
  xT16[((size_t)b * DD + c) * NN + n] = v;
}

// ---------------- Kernel 3: pack weights_pool -> PT[(o,k,i)][d] bf16 ---------
__global__ __launch_bounds__(256) void k_pack_P(const float* __restrict__ wpool,
                                                __bf16* __restrict__ PT) {
  size_t idx = (size_t)blockIdx.x * blockDim.x + threadIdx.x;  // EE*2*DD*DD
  int d = (int)(idx / (2 * DD * DD));
  int rem = (int)(idx % (2 * DD * DD));
  int k = rem / (DD * DD);
  int rem2 = rem % (DD * DD);
  int i = rem2 / DD, o = rem2 % DD;
  size_t j = (size_t)o * KK2 + (size_t)k * DD + i;   // (o,k,i)
  PT[j * EE + d] = (__bf16)wpool[idx];
}

// ---------------- Kernel 4: bias[n][o] = e[n,:] @ bias_pool ------------------
__global__ __launch_bounds__(256) void k_bias(const float* __restrict__ e32,
                                              const float* __restrict__ bpool,
                                              float* __restrict__ bias) {
  int idx = blockIdx.x * blockDim.x + threadIdx.x;   // NN*DD
  int n = idx / DD, o = idx % DD;
  const float* er = e32 + (size_t)n * EE;
  float acc = 0.f;
#pragma unroll
  for (int d = 0; d < EE; ++d) acc = fmaf(er[d], bpool[(size_t)d * DD + o], acc);
  bias[idx] = acc;
}

// ---------------- Kernel 5: adj = softmax_rows(e @ e^T), fused, bf16 out -----
// One workgroup per 16-row block; logits (16 x 2048 f32 = 128KB) in dynamic LDS.
__global__ __launch_bounds__(256) void k_adj(const __bf16* __restrict__ e16,
                                             __bf16* __restrict__ adj) {
  extern __shared__ float lg[];                // [16][NN]
  int tid = threadIdx.x, lane = tid & 31, wave = tid >> 5;
  int nblk = blockIdx.x;                       // 128 blocks
  const __bf16* Abase = e16 + (size_t)nblk * 16 * EE;
  v16bf a0 = load_A16(Abase, EE, 0, lane);
  v16bf a1 = load_A16(Abase, EE, 32, lane);
  int col = lane & 15, hi = lane >> 4;
#pragma unroll 1
  for (int t = 0; t < 16; ++t) {
    int ct = wave + 8 * t;                     // 0..127 column tiles
    const __bf16* Bt = e16 + (size_t)ct * 16 * EE;  // B^T rows are rows of e
    v16bf b0 = load_Bt16(Bt, EE, 0, lane);
    v16bf b1 = load_Bt16(Bt, EE, 32, lane);
    v8f acc = {0.f, 0.f, 0.f, 0.f, 0.f, 0.f, 0.f, 0.f};
    acc = wmma_bf16(a0, b0, acc);
    acc = wmma_bf16(a1, b1, acc);
#pragma unroll
    for (int j = 0; j < 8; ++j) lg[(size_t)(j + hi * 8) * NN + ct * 16 + col] = acc[j];
  }
  __syncthreads();
  // softmax over 2048 columns; 16 threads per row (aligned 16-lane groups)
  int row = tid >> 4, sub = tid & 15;
  const float* lr = lg + (size_t)row * NN;
  float m = -3.4e38f;
  for (int c2 = sub; c2 < NN; c2 += 16) m = fmaxf(m, lr[c2]);
#pragma unroll
  for (int off = 8; off; off >>= 1) m = fmaxf(m, __shfl_xor(m, off, 16));
  float ssum = 0.f;
  for (int c2 = sub; c2 < NN; c2 += 16) ssum += __expf(lr[c2] - m);
#pragma unroll
  for (int off = 8; off; off >>= 1) ssum += __shfl_xor(ssum, off, 16);
  float inv = 1.f / ssum;
  size_t gr = (size_t)(nblk * 16 + row) * NN;
  for (int c2 = sub; c2 < NN; c2 += 16)
    adj[gr + c2] = (__bf16)(__expf(lr[c2] - m) * inv);
}

// ---------------- Kernel 6: x_agg[b] = adj @ x[b]  (bf16 out) ----------------
// grid (64 nblk, 32 batch); wave w: M-subtile w&1, N columns 32*(w>>1)
__global__ __launch_bounds__(256) void k_xagg(const __bf16* __restrict__ adj,
                                              const __bf16* __restrict__ xT16,
                                              __bf16* __restrict__ xagg) {
  int lane = threadIdx.x & 31, wave = threadIdx.x >> 5;
  int nblk = blockIdx.x, b = blockIdx.y;
  int mrow = nblk * 32 + (wave & 1) * 16;
  int c0 = (wave >> 1) * 32, c1 = c0 + 16;
  const __bf16* Abase = adj + (size_t)mrow * NN;
  const __bf16* Bt0 = xT16 + ((size_t)b * DD + c0) * NN;
  const __bf16* Bt1 = xT16 + ((size_t)b * DD + c1) * NN;
  v8f acc0 = {0.f, 0.f, 0.f, 0.f, 0.f, 0.f, 0.f, 0.f};
  v8f acc1 = acc0;
#pragma unroll 1
  for (int k0 = 0; k0 < NN; k0 += 32) {
    v16bf a  = load_A16(Abase, NN, k0, lane);
    v16bf b0 = load_Bt16(Bt0, NN, k0, lane);
    v16bf b1 = load_Bt16(Bt1, NN, k0, lane);
    acc0 = wmma_bf16(a, b0, acc0);
    acc1 = wmma_bf16(a, b1, acc1);
  }
  __bf16* out = xagg + ((size_t)b * NN + mrow) * DD;
  store_C_bf16(out + c0, DD, acc0, lane);
  store_C_bf16(out + c1, DD, acc1, lane);
}

// ---------------- Kernel 7: weight gen (per chunk of 256 nodes) --------------
// WT[nn][j] = sum_d e[n0+nn, d] * PT[j][d],  j = o*256 + k*128 + i
__global__ __launch_bounds__(256) void k_wgen(int n0,
                                              const __bf16* __restrict__ e16,
                                              const __bf16* __restrict__ PT,
                                              __bf16* __restrict__ WT) {
  int lane = threadIdx.x & 31, wave = threadIdx.x >> 5;
  int mtile = blockIdx.x;                       // 16 tiles of 16 nodes
  int grp = blockIdx.y;                         // 32 groups
  const __bf16* Abase = e16 + (size_t)(n0 + mtile * 16) * EE;
  v16bf a0 = load_A16(Abase, EE, 0, lane);
  v16bf a1 = load_A16(Abase, EE, 32, lane);
#pragma unroll 1
  for (int t = 0; t < 8; ++t) {
    int ntile = grp * 64 + wave * 8 + t;        // 0..2047
    size_t j0 = (size_t)ntile * 16;
    const __bf16* Bt = PT + j0 * EE;            // PT rows = B^T rows
    v16bf b0 = load_Bt16(Bt, EE, 0, lane);
    v16bf b1 = load_Bt16(Bt, EE, 32, lane);
    v8f acc = {0.f, 0.f, 0.f, 0.f, 0.f, 0.f, 0.f, 0.f};
    acc = wmma_bf16(a0, b0, acc);
    acc = wmma_bf16(a1, b1, acc);
    store_C_bf16(WT + (size_t)(mtile * 16) * NJ + j0, NJ, acc, lane);
  }
}

// ---------------- Kernel 8: final per-node GEMM + bias (per chunk) -----------
// out[b,n,o] = bias[n,o] + sum_ki Xg[b,n,ki] * W_n[ki,o]
__global__ __launch_bounds__(256) void k_final(int n0,
                                               const __bf16* __restrict__ x16,
                                               const __bf16* __restrict__ xagg,
                                               const __bf16* __restrict__ WT,
                                               const float* __restrict__ bias,
                                               float* __restrict__ out) {
  int lane = threadIdx.x & 31, wave = threadIdx.x >> 5;
  int nn = blockIdx.x, n = n0 + nn;
  int mtile = wave & 1;                          // batch rows 16*mtile..+15
  int o0 = (wave >> 1) * 32, o1 = o0 + 16;
  int col = lane & 15, hi = lane >> 4;
  int bb = mtile * 16 + col;                     // A-row batch index for this lane
  const __bf16* xrow = x16  + ((size_t)bb * NN + n) * DD;
  const __bf16* grow = xagg + ((size_t)bb * NN + n) * DD;
  const __bf16* WTn = WT + (size_t)nn * NJ;
  float bv0 = bias[(size_t)n * DD + o0 + col];
  float bv1 = bias[(size_t)n * DD + o1 + col];
  v8f acc0, acc1;
#pragma unroll
  for (int j = 0; j < 8; ++j) { acc0[j] = bv0; acc1[j] = bv1; }
#pragma unroll 1
  for (int kk = 0; kk < KK2; kk += 32) {
    const __bf16* s = (kk < DD) ? (xrow + kk) : (grow + (kk - DD));
    v8bf lo = *(const v8bf*)(s + hi * 8);
    v8bf up = *(const v8bf*)(s + 16 + hi * 8);
    v16bf a;
#pragma unroll
    for (int i = 0; i < 8; ++i) { a[i] = lo[i]; a[i + 8] = up[i]; }
    v16bf b0 = load_Bt16(WTn + (size_t)o0 * KK2, KK2, kk, lane);
    v16bf b1 = load_Bt16(WTn + (size_t)o1 * KK2, KK2, kk, lane);
    acc0 = wmma_bf16(a, b0, acc0);
    acc1 = wmma_bf16(a, b1, acc1);
  }
#pragma unroll
  for (int j = 0; j < 8; ++j) {
    int br = mtile * 16 + j + hi * 8;
    out[((size_t)br * NN + n) * DD + o0 + col] = acc0[j];
    out[((size_t)br * NN + n) * DD + o1 + col] = acc1[j];
  }
}

extern "C" void kernel_launch(void* const* d_in, const int* in_sizes, int n_in,
                              void* d_out, int out_size, void* d_ws, size_t ws_size,
                              hipStream_t stream) {
  const float* x     = (const float*)d_in[0];   // [32,2048,128]
  const float* nodee = (const float*)d_in[1];   // [2048,64]
  const float* timee = (const float*)d_in[2];   // [64]
  const float* wpool = (const float*)d_in[3];   // [64,2,128,128]
  const float* bpool = (const float*)d_in[4];   // [64,128]
  const float* lnw   = (const float*)d_in[5];   // [64]
  const float* lnb   = (const float*)d_in[6];   // [64]
  float* out = (float*)d_out;                   // [32,2048,128]

  char* ws = (char*)d_ws;
  size_t off = 0;
  float*  e32   = (float*)(ws + off);  off += (size_t)NN * EE * 4;        // 512K
  __bf16* e16   = (__bf16*)(ws + off); off += (size_t)NN * EE * 2;        // 256K
  float*  bias  = (float*)(ws + off);  off += (size_t)NN * DD * 4;        // 1M
  __bf16* PT    = (__bf16*)(ws + off); off += (size_t)NJ * EE * 2;        // 4M
  __bf16* x16   = (__bf16*)(ws + off); off += (size_t)BB * NN * DD * 2;   // 16M
  __bf16* xT16  = (__bf16*)(ws + off); off += (size_t)BB * NN * DD * 2;   // 16M
  __bf16* adj   = (__bf16*)(ws + off); off += (size_t)NN * NN * 2;        // 8M
  __bf16* xagg  = (__bf16*)(ws + off); off += (size_t)BB * NN * DD * 2;   // 16M
  __bf16* WT    = (__bf16*)(ws + off); off += (size_t)CHUNK * NJ * 2;     // 16M
  (void)ws_size; (void)in_sizes; (void)n_in; (void)out_size;

  k_prep_e<<<NN / 8, 256, 0, stream>>>(nodee, timee, lnw, lnb, e32, e16);
  k_pack_x<<<(BB * NN * DD) / 256, 256, 0, stream>>>(x, x16, xT16);
  k_pack_P<<<(EE * 2 * DD * DD) / 256, 256, 0, stream>>>(wpool, PT);
  k_bias<<<(NN * DD) / 256, 256, 0, stream>>>(e32, bpool, bias);
  k_adj<<<NN / 16, 256, 16 * NN * sizeof(float), stream>>>(e16, adj);
  k_xagg<<<dim3(NN / 32, BB), 256, 0, stream>>>(adj, xT16, xagg);
  for (int c = 0; c < NN / CHUNK; ++c) {
    int n0 = c * CHUNK;
    k_wgen<<<dim3(CHUNK / 16, 32), 256, 0, stream>>>(n0, e16, PT, WT);
    k_final<<<CHUNK, 256, 0, stream>>>(n0, x16, xagg, WT, bias, out);
  }
}
